// Model_67697274520270
// MI455X (gfx1250) — compile-verified, matrix operand order checked by reading, not run on hardware
//
#include <hip/hip_runtime.h>
#include <stdint.h>
#include <stddef.h>

// ---------------- types ----------------
typedef __bf16 bf16_t;
typedef bf16_t bf16x16 __attribute__((ext_vector_type(16)));
typedef float  f32x8   __attribute__((ext_vector_type(8)));

union AFrag { bf16x16 v; unsigned short u[16]; };
union CFrag { f32x8 v; float f[8]; };

__device__ __forceinline__ float bf2f(unsigned short u) {
    union { float f; unsigned int i; } x; x.i = ((unsigned int)u) << 16; return x.f;
}
__device__ __forceinline__ unsigned short f2bf(float f) {
    union { float f; unsigned int i; } x; x.f = f;
    unsigned int b = x.i;
    unsigned int r = (b + 0x7FFFu + ((b >> 16) & 1u)) >> 16;   // RNE
    return (unsigned short)r;
}

// ---------------- generic implicit-GEMM conv (bf16 WMMA) ----------------
// Block: 256 threads (8 waves). Tile: M=128 x N=64. Each wave: 16 rows x 64 cols
// (4 WMMA accumulators), 4 v_wmma_f32_16x16x32_bf16 per K-chunk.
// B tile staged with global_load_async_to_lds_b128 (weights pre-padded to [Co][Ktot]).
#define F_RELU_IN  1
#define F_ADD_RES  2
#define F_RELU_OUT 4
#define F_OUT_F32  8

struct ConvP {
    const unsigned short* x;      // input, NCHW bf16
    const unsigned short* w;      // weights [Co][Ktot] bf16, K-padded (pre-flipped for conv-T)
    const float* bias;            // nullable, f32
    const unsigned short* res;    // nullable residual (output shape)
    unsigned short* y;            // bf16 out
    float* yf;                    // f32 out (when F_OUT_F32)
    int N, Ci, H, W, Co, Ho, Wo;
    int SH, SW, PH, PW, DH, DW;   // DH/DW = lhs dilation (conv-T stride)
    int M, Ktot, flags;
};

template <int KHW, int KWc>
__launch_bounds__(256)
__global__ void conv_wmma(ConvP p) {
    __shared__ unsigned short As[128 * 33];                 // [row][k]
    __shared__ __align__(16) unsigned short Bs[64][40];     // [n][k], 80B pitch (16B-aligned segs)
    const int tid  = threadIdx.x;
    const int lane = tid & 31;
    const int wv   = tid >> 5;
    const int m0   = blockIdx.x * 128;
    const int co0  = blockIdx.y * 64;
    const int Kreal = p.Ci * KHW;
    const int HW   = p.H * p.W;

    // --- hoisted per-thread A-stage row decomposition (row fixed per thread) ---
    const int amr = tid & 127;          // row this thread stages
    const int akg = tid >> 7;           // k-slot group (0/1): k = akg + 2*i
    const int am  = m0 + amr;
    const bool amok = am < p.M;
    int hbase = 0, wbase = 0; size_t nbase = 0;
    if (amok) {
        int wo = am % p.Wo; int t = am / p.Wo;
        int ho = t % p.Ho;  int n = t / p.Ho;
        hbase = ho * p.SH - p.PH;
        wbase = wo * p.SW - p.PW;
        nbase = (size_t)n * p.Ci * HW;
    }
    // --- B-stage: one async b128 per thread per chunk (16B, contiguous in K) ---
    const int bnn  = tid >> 2;          // column 0..63
    const int bseg = tid & 3;           // 16B segment within the 64B K-chunk
    const int bco  = co0 + bnn;
    const bool bok = bco < p.Co;
    const unsigned short* bsrc = p.w + (size_t)bco * p.Ktot + bseg * 8;
    const unsigned blds = (unsigned)(uintptr_t)&Bs[bnn][bseg * 8];  // flat->LDS: low 32 bits

    CFrag c[4];
#pragma unroll
    for (int s = 0; s < 4; s++)
#pragma unroll
        for (int i = 0; i < 8; i++) c[s].f[i] = 0.f;

    for (int kb = 0; kb < p.Ktot; kb += 32) {
        // ---- async-stage B tile: 32 K x 64 cols (ASYNCcnt DMA path) ----
        if (bok) {
            asm volatile("global_load_async_to_lds_b128 %0, %1, off"
                         :: "v"(blds), "v"(bsrc + kb) : "memory");
        }
        // ---- stage A tile: 128 rows x 32 K (im2col on the fly) ----
#pragma unroll 4
        for (int i = 0; i < 16; i++) {
            const int k = akg + 2 * i;
            const int kk = kb + k;
            unsigned short v = 0;
            if (amok && kk < Kreal) {
                int ci, kh, kw;
                if (KHW == 1) { ci = kk; kh = 0; kw = 0; }
                else {
                    ci = kk / KHW; int rr = kk - ci * KHW;
                    kh = rr / KWc; kw = rr - kh * KWc;
                }
                int ht = hbase + kh;
                int wt = wbase + kw;
                bool ok = (ht >= 0) && (wt >= 0);
                int hi = ht, wi = wt;
                if (p.DH > 1) { ok = ok && (ht % p.DH == 0); hi = ht / p.DH; }
                if (p.DW > 1) { ok = ok && (wt % p.DW == 0); wi = wt / p.DW; }
                ok = ok && (hi < p.H) && (wi < p.W);
                if (ok) {
                    v = p.x[nbase + (size_t)ci * HW + (size_t)hi * p.W + wi];
                    if ((p.flags & F_RELU_IN) && (v & 0x8000u)) v = 0;
                }
            }
            As[amr * 33 + k] = v;
        }
        asm volatile("s_wait_asynccnt 0x0" ::: "memory");
        __syncthreads();

        // ---- fragments per CDNA5 16-bit A/B layout:
        // lanes 0-15: K {0..7,16..23}; lanes 16-31: K {8..15,24..31}
        const int rc = lane & 15;
        const int k0 = (lane >> 4) * 8;
        AFrag a;
        const unsigned short* ap = &As[(wv * 16 + rc) * 33];
#pragma unroll
        for (int i = 0; i < 8; i++) { a.u[i] = ap[k0 + i]; a.u[8 + i] = ap[16 + k0 + i]; }
#pragma unroll
        for (int s = 0; s < 4; s++) {
            AFrag b;
            const unsigned short* bp = &Bs[s * 16 + rc][0];
#pragma unroll
            for (int i = 0; i < 8; i++) {
                b.u[i]     = bp[k0 + i];
                b.u[8 + i] = bp[16 + k0 + i];
            }
            c[s].v = __builtin_amdgcn_wmma_f32_16x16x32_bf16(false, a.v, false, b.v,
                                                             (short)0, c[s].v, false, false);
        }
        __syncthreads();
    }

    // ---- epilogue: C layout = VGPR r -> row (lanes>=16: +8), lane&15 -> col ----
    const int cc = lane & 15;
    const int rbase = (lane >> 4) * 8;
#pragma unroll
    for (int r = 0; r < 8; r++) {
        const int m = m0 + wv * 16 + rbase + r;
        if (m >= p.M) continue;
        int wo = m % p.Wo; int t = m / p.Wo;
        int ho = t % p.Ho; int n = t / p.Ho;
        const size_t obase = ((size_t)n * p.Co * p.Ho + (size_t)ho) * p.Wo + wo;
#pragma unroll
        for (int s = 0; s < 4; s++) {
            const int col = co0 + s * 16 + cc;
            if (col < p.Co) {
                float v = c[s].f[r];
                if (p.bias) v += p.bias[col];
                const size_t oi = obase + (size_t)col * p.Ho * p.Wo;
                if (p.flags & F_ADD_RES) v += bf2f(p.res[oi]);
                if (p.flags & F_RELU_OUT) v = v > 0.f ? v : 0.f;
                if (p.flags & F_OUT_F32) p.yf[oi] = v; else p.y[oi] = f2bf(v);
            }
        }
    }
}

// ---------------- VQ: argmin over codebook via WMMA ----------------
// z: (16,256,4,64) NCHW bf16; flat row r=((n*4+h)*64+w), col=c. 4096 rows.
__launch_bounds__(32)
__global__ void vq_argmin(const unsigned short* z, const unsigned short* Ebf,
                          const float* Esq, int* idxOut) {
    __shared__ unsigned short zt[16 * 257];
    const int lane = threadIdx.x;
    const int r0 = blockIdx.x * 16;
    for (int e = lane; e < 16 * 256; e += 32) {
        int r = e >> 8, cc = e & 255;
        int gr = r0 + r;
        int n = gr >> 8; int rem = gr & 255; int h = rem >> 6; int w = rem & 63;
        zt[r * 257 + cc] = z[((size_t)(n * 256 + cc) * 4 + h) * 64 + w];
    }
    __syncthreads();

    float best[8]; int bidx[8];
#pragma unroll
    for (int i = 0; i < 8; i++) { best[i] = 3.4e38f; bidx[i] = 0; }
    const int rc = lane & 15;
    const int k0 = (lane >> 4) * 8;

    for (int jt = 0; jt < 32; jt++) {
        CFrag c8;
#pragma unroll
        for (int i = 0; i < 8; i++) c8.f[i] = 0.f;
        const int j = jt * 16 + rc;
        for (int kc = 0; kc < 256; kc += 32) {
            AFrag a, b;
            const unsigned short* ap = &zt[rc * 257 + kc];
#pragma unroll
            for (int i = 0; i < 8; i++) { a.u[i] = ap[k0 + i]; a.u[8 + i] = ap[16 + k0 + i]; }
            const unsigned short* bp = &Ebf[(size_t)j * 256 + kc];
#pragma unroll
            for (int i = 0; i < 8; i++) { b.u[i] = bp[k0 + i]; b.u[8 + i] = bp[16 + k0 + i]; }
            c8.v = __builtin_amdgcn_wmma_f32_16x16x32_bf16(false, a.v, false, b.v,
                                                           (short)0, c8.v, false, false);
        }
        float ej = Esq[j];
#pragma unroll
        for (int r = 0; r < 8; r++) {
            float val = ej - 2.0f * c8.f[r];          // ||E||^2 - 2 z.E  (argmin-equivalent)
            if (val < best[r]) { best[r] = val; bidx[r] = j; }
        }
    }
#pragma unroll
    for (int r = 0; r < 8; r++) {
        float v = best[r]; int ix = bidx[r];
        for (int mk = 1; mk < 16; mk <<= 1) {
            float ov = __shfl_xor(v, mk, 32);
            int   oi = __shfl_xor(ix, mk, 32);
            if (ov < v || (ov == v && oi < ix)) { v = ov; ix = oi; }
        }
        best[r] = v; bidx[r] = ix;
    }
    if (lane == 0)  { for (int r = 0; r < 8; r++) idxOut[r0 + r]     = bidx[r]; }
    if (lane == 16) { for (int r = 0; r < 8; r++) idxOut[r0 + 8 + r] = bidx[r]; }
}

// ---------------- small helper kernels ----------------
__global__ void f2bf_kernel(const float* s, unsigned short* d, int n) {
    int i = blockIdx.x * 256 + threadIdx.x; if (i < n) d[i] = f2bf(s[i]);
}
// Conv weight (Co,Ci,KH,KW) f32 -> [Co][Ktot] bf16, zero-padded K
__global__ void convw_prep(const float* s, unsigned short* d, int Co, int Kreal, int Ktot) {
    int i = blockIdx.x * 256 + threadIdx.x;
    int tot = Co * Ktot; if (i >= tot) return;
    int co = i / Ktot; int kk = i - co * Ktot;
    d[i] = (kk < Kreal) ? f2bf(s[(size_t)co * Kreal + kk]) : (unsigned short)0;
}
// ConvTranspose2d weight (I,O,KH,KW) f32 -> [O][Ktot] bf16, transposed+flipped, zero-padded K
__global__ void convtw_prep(const float* s, unsigned short* d, int I, int O,
                            int KH, int KW, int Ktot) {
    int i = blockIdx.x * 256 + threadIdx.x;
    int tot = O * Ktot; if (i >= tot) return;
    int o = i / Ktot; int kk = i - o * Ktot;
    int khw = KH * KW; int Kreal = I * khw;
    unsigned short v = 0;
    if (kk < Kreal) {
        int ci = kk / khw; int rr = kk - ci * khw;
        int kh = rr / KW; int kw = rr - kh * KW;
        v = f2bf(s[(((size_t)ci * O + o) * KH + (KH - 1 - kh)) * KW + (KW - 1 - kw)]);
    }
    d[i] = v;
}
__global__ void esq_kernel(const float* E, float* Esq) {
    int j = blockIdx.x * 128 + threadIdx.x; if (j >= 512) return;
    float s = 0.f;
    for (int d0 = 0; d0 < 256; d0++) { float v = E[j * 256 + d0]; s += v * v; }
    Esq[j] = s;
}
__global__ void zero_kernel(float* p, int n) {
    int i = blockIdx.x * 256 + threadIdx.x; if (i < n) p[i] = 0.f;
}
// gather q = E[idx], build decoder input (NCHW bf16), accumulate sum((q-z)^2)
__global__ void gather_loss(const int* idx, const unsigned short* Ebf,
                            const unsigned short* z, unsigned short* decin, float* lossAcc) {
    int i = blockIdx.x * 256 + threadIdx.x;
    float d = 0.f;
    if (i < 4096 * 256) {
        int cc = i & 255; int r = i >> 8;
        int n = r >> 8; int rem = r & 255; int h = rem >> 6; int w = rem & 63;
        unsigned short q = Ebf[(size_t)idx[r] * 256 + cc];
        size_t zi = ((size_t)(n * 256 + cc) * 4 + h) * 64 + w;
        unsigned short zz = z[zi];
        decin[zi] = q;
        float df = bf2f(q) - bf2f(zz);
        d = df * df;
    }
    __shared__ float sm[256];
    sm[threadIdx.x] = d; __syncthreads();
    for (int s = 128; s > 0; s >>= 1) { if (threadIdx.x < s) sm[threadIdx.x] += sm[threadIdx.x + s]; __syncthreads(); }
    if (threadIdx.x == 0) atomicAdd(lossAcc, sm[0]);
}
__global__ void hist_kernel(const int* idx, float* counts) {
    int i = blockIdx.x * 256 + threadIdx.x; if (i < 4096) atomicAdd(&counts[idx[i]], 1.0f);
}
__global__ void finalize_kernel(const float* lossAcc, const float* counts,
                                float* out_loss, float* out_perp) {
    __shared__ float sm[512];
    int t = threadIdx.x;
    float p = counts[t] * (1.0f / 4096.0f);
    sm[t] = p * logf(p + 1e-10f);
    __syncthreads();
    for (int s = 256; s > 0; s >>= 1) { if (t < s) sm[t] += sm[t + s]; __syncthreads(); }
    if (t == 0) {
        *out_perp = expf(-sm[0]);
        *out_loss = 1.25f * lossAcc[0] / (4096.0f * 256.0f);  // (1+CC)*mse
    }
}

// ---------------- host orchestration ----------------
extern "C" void kernel_launch(void* const* d_in, const int* in_sizes, int n_in,
                              void* d_out, int out_size, void* d_ws, size_t ws_size,
                              hipStream_t stream) {
    (void)in_sizes; (void)n_in; (void)out_size; (void)ws_size;
    char* base = (char*)d_ws;
    size_t off = 0;
    auto alloc = [&](size_t bytes) -> void* {
        void* p = base + off; off = (off + bytes + 255) & ~(size_t)255; return p;
    };
    float* lossAcc = (float*)alloc(4);
    float* counts  = (float*)alloc(512 * 4);
    int*   idxbuf  = (int*)alloc(4096 * 4);
    float* Esq     = (float*)alloc(512 * 4);
    unsigned short* X0  = (unsigned short*)alloc((size_t)4194304 * 2);
    unsigned short* X1  = (unsigned short*)alloc((size_t)4194304 * 2);
    unsigned short* HID = (unsigned short*)alloc((size_t)33554432 * 2);

    auto cvt = [&](int di, size_t n) -> unsigned short* {
        unsigned short* p = (unsigned short*)alloc(n * 2);
        f2bf_kernel<<<(unsigned)((n + 255) / 256), 256, 0, stream>>>((const float*)d_in[di], p, (int)n);
        return p;
    };
    auto prepW = [&](int di, int Co, int Ci, int KH, int KW) -> unsigned short* {
        int Kreal = Ci * KH * KW, Kt = (Kreal + 31) & ~31;
        size_t n = (size_t)Co * Kt;
        unsigned short* p = (unsigned short*)alloc(n * 2);
        convw_prep<<<(unsigned)((n + 255) / 256), 256, 0, stream>>>((const float*)d_in[di], p, Co, Kreal, Kt);
        return p;
    };
    auto prepWT = [&](int di, int I, int O, int KH, int KW) -> unsigned short* {
        int Kreal = I * KH * KW, Kt = (Kreal + 31) & ~31;
        size_t n = (size_t)O * Kt;
        unsigned short* p = (unsigned short*)alloc(n * 2);
        convtw_prep<<<(unsigned)((n + 255) / 256), 256, 0, stream>>>((const float*)d_in[di], p, I, O, KH, KW, Kt);
        return p;
    };
    auto conv = [&](const unsigned short* xin, const unsigned short* w, const float* bias,
                    const unsigned short* res, unsigned short* y, float* yf,
                    int N, int Ci, int H, int W, int Co, int Ho, int Wo,
                    int KH, int KW, int SH, int SW, int PH, int PW, int DH, int DW, int flags) {
        ConvP p;
        p.x = xin; p.w = w; p.bias = bias; p.res = res; p.y = y; p.yf = yf;
        p.N = N; p.Ci = Ci; p.H = H; p.W = W; p.Co = Co; p.Ho = Ho; p.Wo = Wo;
        p.SH = SH; p.SW = SW; p.PH = PH; p.PW = PW; p.DH = DH; p.DW = DW;
        p.M = N * Ho * Wo;
        p.Ktot = ((Ci * KH * KW) + 31) & ~31;
        p.flags = flags;
        dim3 g((p.M + 127) / 128, (Co + 63) / 64);
        const int khw = KH * KW;
        if      (khw == 1)  conv_wmma<1, 1><<<g, 256, 0, stream>>>(p);
        else if (khw == 9)  conv_wmma<9, 3><<<g, 256, 0, stream>>>(p);
        else if (khw == 16) conv_wmma<16, 4><<<g, 256, 0, stream>>>(p);
        else if (khw == 4)  conv_wmma<4, 2><<<g, 256, 0, stream>>>(p);
        else                conv_wmma<6, 3><<<g, 256, 0, stream>>>(p);   // 2x3
    };

    // ---- weight prep (input flatten order: x, enc[0..5]{w,b,s0w1,s0w2,s1w1,s1w2},
    //      dec_conv1{w,b}, dec_ct[0..4]{w,b}, dec_stacks[0..4]{w1,w2,w1,w2}, codebook)
    static const int eci[6] = {1, 4, 16, 32, 64, 128};
    static const int eco[6] = {4, 16, 32, 64, 128, 256};
    static const int ekh[6] = {4, 4, 3, 3, 3, 3};
    static const int esh[6] = {2, 2, 2, 2, 1, 1};
    static const int esw[6] = {2, 2, 2, 1, 1, 1};
    static const int eH[7]  = {64, 32, 16, 8, 4, 4, 4};
    static const int eW[7]  = {512, 256, 128, 64, 64, 64, 64};

    unsigned short* ew[6]; const float* eb[6];
    unsigned short* ew1[6][2]; unsigned short* ew2[6][2];
    for (int i = 0; i < 6; i++) {
        int bi = 1 + i * 6;
        ew[i] = prepW(bi, eco[i], eci[i], ekh[i], ekh[i]);
        eb[i] = (const float*)d_in[bi + 1];
        ew1[i][0] = prepW(bi + 2, 256, eco[i], 3, 3);
        ew2[i][0] = prepW(bi + 3, eco[i], 256, 1, 1);
        ew1[i][1] = prepW(bi + 4, 256, eco[i], 3, 3);
        ew2[i][1] = prepW(bi + 5, eco[i], 256, 1, 1);
    }
    unsigned short* dc1w = prepW(37, 128, 256, 3, 3);
    const float* dc1b = (const float*)d_in[38];

    static const int cti[5] = {128, 64, 32, 16, 4};
    static const int cto[5] = {64, 32, 16, 4, 1};
    static const int ctKH[5] = {2, 2, 2, 2, 3};
    static const int ctKW[5] = {2, 2, 2, 3, 3};
    static const int ctDH[5] = {2, 2, 2, 2, 1};
    static const int ctDW[5] = {2, 2, 2, 1, 1};
    static const int ctHin[5]  = {4, 8, 16, 32, 64};
    static const int ctWin[5]  = {64, 128, 256, 512, 512};
    static const int ctHout[5] = {8, 16, 32, 64, 64};
    static const int ctWout[5] = {128, 256, 512, 512, 512};
    unsigned short* ctw[5]; const float* ctb[5];
    for (int j = 0; j < 5; j++) {
        ctw[j] = prepWT(39 + 2 * j, cti[j], cto[j], ctKH[j], ctKW[j]);
        ctb[j] = (const float*)d_in[40 + 2 * j];
    }
    static const int dch[5] = {128, 64, 32, 16, 4};
    unsigned short* dw1[5][2]; unsigned short* dw2[5][2];
    for (int s = 0; s < 5; s++) {
        int bi = 49 + s * 4;
        dw1[s][0] = prepW(bi + 0, dch[s], dch[s], 3, 3);
        dw2[s][0] = prepW(bi + 1, dch[s], dch[s], 1, 1);
        dw1[s][1] = prepW(bi + 2, dch[s], dch[s], 3, 3);
        dw2[s][1] = prepW(bi + 3, dch[s], dch[s], 1, 1);
    }
    unsigned short* Ebf = cvt(69, (size_t)512 * 256);
    esq_kernel<<<4, 128, 0, stream>>>((const float*)d_in[69], Esq);
    zero_kernel<<<1, 256, 0, stream>>>(lossAcc, 1);
    zero_kernel<<<2, 256, 0, stream>>>(counts, 512);

    // ---- encoder ----
    f2bf_kernel<<<(524288 + 255) / 256, 256, 0, stream>>>((const float*)d_in[0], X0, 524288);
    unsigned short* cur = X0; unsigned short* other = X1;

    auto res_stack = [&](unsigned short* w1a, unsigned short* w2a,
                         unsigned short* w1b, unsigned short* w2b,
                         int C, int RH, int H, int W) {
        conv(cur, w1a, nullptr, nullptr, HID, nullptr, 16, C, H, W, RH, H, W,
             3, 3, 1, 1, 1, 1, 1, 1, F_RELU_IN);
        conv(HID, w2a, nullptr, cur, other, nullptr, 16, RH, H, W, C, H, W,
             1, 1, 1, 1, 0, 0, 1, 1, F_RELU_IN | F_ADD_RES);
        unsigned short* t = cur; cur = other; other = t;
        conv(cur, w1b, nullptr, nullptr, HID, nullptr, 16, C, H, W, RH, H, W,
             3, 3, 1, 1, 1, 1, 1, 1, F_RELU_IN);
        conv(HID, w2b, nullptr, cur, other, nullptr, 16, RH, H, W, C, H, W,
             1, 1, 1, 1, 0, 0, 1, 1, F_RELU_IN | F_ADD_RES | F_RELU_OUT);
        t = cur; cur = other; other = t;
    };

    for (int i = 0; i < 6; i++) {
        conv(cur, ew[i], eb[i], nullptr, other, nullptr,
             16, eci[i], eH[i], eW[i], eco[i], eH[i + 1], eW[i + 1],
             ekh[i], ekh[i], esh[i], esw[i], 1, 1, 1, 1, 0);
        unsigned short* t = cur; cur = other; other = t;
        res_stack(ew1[i][0], ew2[i][0], ew1[i][1], ew2[i][1], eco[i], 256, eH[i + 1], eW[i + 1]);
    }

    // ---- vector quantizer ----
    vq_argmin<<<256, 32, 0, stream>>>(cur, Ebf, Esq, idxbuf);
    gather_loss<<<4096, 256, 0, stream>>>(idxbuf, Ebf, cur, other, lossAcc);
    hist_kernel<<<16, 256, 0, stream>>>(idxbuf, counts);
    { unsigned short* t = cur; cur = other; other = t; }   // cur = decoder input (q)

    // ---- decoder ----
    conv(cur, dc1w, dc1b, nullptr, other, nullptr, 16, 256, 4, 64, 128, 4, 64,
         3, 3, 1, 1, 1, 1, 1, 1, 0);
    { unsigned short* t = cur; cur = other; other = t; }
    res_stack(dw1[0][0], dw2[0][0], dw1[0][1], dw2[0][1], 128, 128, 4, 64);

    for (int j = 0; j < 4; j++) {
        conv(cur, ctw[j], ctb[j], nullptr, other, nullptr,
             16, cti[j], ctHin[j], ctWin[j], cto[j], ctHout[j], ctWout[j],
             ctKH[j], ctKW[j], 1, 1, /*PH=KH-1-p_h*/ 1, /*PW=KW-1-p_w*/ 1, ctDH[j], ctDW[j], 0);
        unsigned short* t = cur; cur = other; other = t;
        res_stack(dw1[j + 1][0], dw2[j + 1][0], dw1[j + 1][1], dw2[j + 1][1],
                  cto[j], cto[j], ctHout[j], ctWout[j]);
    }
    // final conv-T (4->1, k3 s1 p1 => plain conv with flipped kernel), f32 out into d_out[1..]
    conv(cur, ctw[4], ctb[4], nullptr, nullptr, (float*)d_out + 1,
         16, 4, 64, 512, 1, 64, 512, 3, 3, 1, 1, 1, 1, 1, 1, F_OUT_F32);

    // ---- loss + perplexity ----
    finalize_kernel<<<1, 512, 0, stream>>>(lossAcc, counts,
                                           (float*)d_out, (float*)d_out + 1 + 524288);
}
